// CrossAttention_72009421685099
// MI455X (gfx1250) — matrix operand order repaired and checked by reading, not compiled, not run on hardware
//
#include <hip/hip_runtime.h>
#include <hip/hip_bf16.h>

#define EDIM 1024
#define BSZ  8
#define QLEN 2048
#define KVLEN 2048

typedef __bf16 bf16;
typedef __attribute__((ext_vector_type(16))) __bf16 v16bf;
typedef __attribute__((ext_vector_type(8)))  __bf16 v8bf;
typedef __attribute__((ext_vector_type(4)))  __bf16 v4bf;
typedef __attribute__((ext_vector_type(8)))  float  v8f;

__device__ __forceinline__ bf16 f2bf(float f) {
  union { float f; unsigned u; } x; x.f = f;
  unsigned r = x.u + 0x7fffu + ((x.u >> 16) & 1u);   // round-to-nearest-even
  unsigned short h = (unsigned short)(r >> 16);
  union { unsigned short s; bf16 b; } y; y.s = h;
  return y.b;
}

__device__ __forceinline__ v8f zero_v8f() {
  v8f z;
#pragma unroll
  for (int i = 0; i < 8; ++i) z[i] = 0.0f;
  return z;
}

__device__ __forceinline__ v8f wmma_bf16(v16bf a, v16bf b, v8f c) {
  // D = A(16x32 bf16) * B(32x16 bf16) + C(16x16 f32)
  return __builtin_amdgcn_wmma_f32_16x16x32_bf16(false, a, false, b, (short)0, c,
                                                 false, false);
}

// ---- NT fragment loaders (both operands row-major along K) ----
// A 16x32 bf16: lanes 0-15 row M=lane, K in [c,c+8)+[c+16,c+24), c=0;
// lanes 16-31 same rows, c=8.
__device__ __forceinline__ v16bf load_a_bf16(const bf16* A, int lda, int row0,
                                             int k0, int lane) {
  const bf16* p = A + (size_t)(row0 + (lane & 15)) * lda + (k0 + ((lane >> 4) << 3));
  v8bf lo = *(const v8bf*)p;         // K = c .. c+7
  v8bf hi = *(const v8bf*)(p + 16);  // K = c+16 .. c+23
  v16bf o;
#pragma unroll
  for (int i = 0; i < 8; ++i) { o[i] = lo[i]; o[8 + i] = hi[i]; }
  return o;
}

// B 32x16 bf16: lane (l&15) = column N; K in [c,c+8)+[c+16,c+24), c=8*(lane>=16).
// For NT gemm, column n of B == row n of a row-major [N x K] matrix.
__device__ __forceinline__ v16bf load_b_bf16(const bf16* Bm, int ldb, int n0,
                                             int k0, int lane) {
  const bf16* p = Bm + (size_t)(n0 + (lane & 15)) * ldb + (k0 + ((lane >> 4) << 3));
  v8bf lo = *(const v8bf*)p;
  v8bf hi = *(const v8bf*)(p + 16);
  v16bf o;
#pragma unroll
  for (int i = 0; i < 8; ++i) { o[i] = lo[i]; o[8 + i] = hi[i]; }
  return o;
}

// ---------------- fp32 -> bf16 convert (4-wide) ----------------
__global__ void cvt_f32_bf16(const float* __restrict__ src, bf16* __restrict__ dst,
                             int n) {
  int i = (blockIdx.x * blockDim.x + threadIdx.x) * 4;
  if (i < n) {
    float4 f = *(const float4*)(src + i);
    v4bf o;
    o[0] = f2bf(f.x); o[1] = f2bf(f.y); o[2] = f2bf(f.z); o[3] = f2bf(f.w);
    *(v4bf*)(dst + i) = o;
  }
}

// ---------------- generic NT GEMM: D = A * B^T (+bias) ----------------
// A: [M x K] row-major bf16.  B: [N x K] row-major bf16, optionally batched
// (batch = block_row*128 / rows_per_batch).
// mode 0: bf16 row-major out; mode 1: bf16 per-batch transposed out
// (D[b][n][m_local], builds V^T, packed 16B stores); mode 2: f32 row-major out.
__global__ __launch_bounds__(256) void gemm_nt(
    const bf16* __restrict__ A, int lda,
    const bf16* __restrict__ Bm, long long bstride, int ldb,
    const float* __restrict__ bias, void* __restrict__ Dv, int mode,
    int N, int K, int rows_per_batch) {
  const int lane = threadIdx.x & 31;
  const int wave = threadIdx.x >> 5;
  const int m0 = blockIdx.y * 128 + (wave >> 1) * 32;  // 4 waves along M
  const int n0 = blockIdx.x * 128 + (wave & 1) * 64;   // 2 waves along N
  const int batch = (blockIdx.y * 128) / rows_per_batch;
  const bf16* Bb = Bm + (size_t)batch * (size_t)bstride;

  v8f acc[2][4];
#pragma unroll
  for (int i = 0; i < 2; ++i)
#pragma unroll
    for (int j = 0; j < 4; ++j) acc[i][j] = zero_v8f();

  for (int k = 0; k < K; k += 32) {
    v16bf a0 = load_a_bf16(A, lda, m0, k, lane);
    v16bf a1 = load_a_bf16(A, lda, m0 + 16, k, lane);
    v16bf bb[4];
#pragma unroll
    for (int j = 0; j < 4; ++j) bb[j] = load_b_bf16(Bb, ldb, n0 + 16 * j, k, lane);
#pragma unroll
    for (int j = 0; j < 4; ++j) {
      acc[0][j] = wmma_bf16(a0, bb[j], acc[0][j]);
      acc[1][j] = wmma_bf16(a1, bb[j], acc[1][j]);
    }
  }

  // C/D layout: VGPR r -> (M = r + 8*(lane>=16), N = lane&15)
#pragma unroll
  for (int i = 0; i < 2; ++i) {
#pragma unroll
    for (int j = 0; j < 4; ++j) {
      int n = n0 + 16 * j + (lane & 15);
      float bv = bias ? bias[n] : 0.0f;
      int rbase = m0 + 16 * i + ((lane >> 4) << 3);
      if (mode == 1) {
        // 8 regs -> 8 consecutive m_local: one packed 16B store
        int b = rbase / rows_per_batch;
        int ml = rbase - b * rows_per_batch;
        v8bf pk;
#pragma unroll
        for (int reg = 0; reg < 8; ++reg) pk[reg] = f2bf(acc[i][j][reg] + bv);
        *(v8bf*)&((bf16*)Dv)[(size_t)b * N * rows_per_batch +
                             (size_t)n * rows_per_batch + ml] = pk;
      } else if (mode == 0) {
#pragma unroll
        for (int reg = 0; reg < 8; ++reg)
          ((bf16*)Dv)[(size_t)(rbase + reg) * N + n] = f2bf(acc[i][j][reg] + bv);
      } else {
#pragma unroll
        for (int reg = 0; reg < 8; ++reg)
          ((float*)Dv)[(size_t)(rbase + reg) * N + n] = acc[i][j][reg] + bv;
      }
    }
  }
}

// ---------------- fused scores + mask + softmax ----------------
// Block = (batch, 32 q-rows). Full 32 x 2048 f32 score tile lives in LDS
// (256 KB of the 320 KB/WGP). 8 waves each own a 32x256 strip of S via WMMA
// (4 chunks of 2Mx4N register tiles, same shape as gemm_nt).
__global__ __launch_bounds__(256) void attn_scores_softmax(
    const bf16* __restrict__ Q, const bf16* __restrict__ Kmat,
    const int* __restrict__ mask, bf16* __restrict__ P, float scale) {
  extern __shared__ char smem[];
  float* S   = (float*)smem;                               // [32][KVLEN]
  float* red = (float*)(smem + (size_t)32 * KVLEN * 4);    // [32][8]

  const int b  = blockIdx.y;
  const int q0 = blockIdx.x * 32;
  const bf16* Qb = Q    + (size_t)b * QLEN  * EDIM;
  const bf16* Kb = Kmat + (size_t)b * KVLEN * EDIM;
  const int lane = threadIdx.x & 31;
  const int wave = threadIdx.x >> 5;

  // ---- S = scale * q @ k^T into LDS ----
  for (int chunk = 0; chunk < 4; ++chunk) {
    const int n0 = wave * 256 + chunk * 64;
    v8f acc[2][4];
#pragma unroll
    for (int i = 0; i < 2; ++i)
#pragma unroll
      for (int j = 0; j < 4; ++j) acc[i][j] = zero_v8f();

#pragma unroll 4
    for (int k = 0; k < EDIM; k += 32) {
      v16bf a0 = load_a_bf16(Qb, EDIM, q0, k, lane);
      v16bf a1 = load_a_bf16(Qb, EDIM, q0 + 16, k, lane);
      v16bf bb[4];
#pragma unroll
      for (int j = 0; j < 4; ++j) bb[j] = load_b_bf16(Kb, EDIM, n0 + 16 * j, k, lane);
#pragma unroll
      for (int j = 0; j < 4; ++j) {
        acc[0][j] = wmma_bf16(a0, bb[j], acc[0][j]);
        acc[1][j] = wmma_bf16(a1, bb[j], acc[1][j]);
      }
    }
#pragma unroll
    for (int i = 0; i < 2; ++i)
#pragma unroll
      for (int j = 0; j < 4; ++j) {
        int col = n0 + 16 * j + (lane & 15);
        int rb  = 16 * i + ((lane >> 4) << 3);
#pragma unroll
        for (int reg = 0; reg < 8; ++reg)
          S[(size_t)(rb + reg) * KVLEN + col] = acc[i][j][reg] * scale;
      }
  }
  __syncthreads();

  // ---- masked softmax; 8 threads per row, 4-wide vector accesses ----
  const int row = threadIdx.x >> 3;
  const int sub = threadIdx.x & 7;
  float* Srow = S + (size_t)row * KVLEN;
  const int* mrow = mask + ((size_t)b * QLEN + q0 + row) * KVLEN;
  const float NINF = -__builtin_inff();

  float lmax = NINF;
  for (int i = 0; i < KVLEN / 32; ++i) {
    int c = (i * 8 + sub) * 4;
    int4   m = *(const int4*)(mrow + c);
    float4 s = *(float4*)(Srow + c);
    if (m.x == 0) s.x = NINF;
    if (m.y == 0) s.y = NINF;
    if (m.z == 0) s.z = NINF;
    if (m.w == 0) s.w = NINF;
    *(float4*)(Srow + c) = s;
    lmax = fmaxf(lmax, fmaxf(fmaxf(s.x, s.y), fmaxf(s.z, s.w)));
  }
  red[row * 8 + sub] = lmax;
  __syncthreads();
  float rmax = red[row * 8];
#pragma unroll
  for (int j = 1; j < 8; ++j) rmax = fmaxf(rmax, red[row * 8 + j]);
  __syncthreads();

  float lsum = 0.0f;
  for (int i = 0; i < KVLEN / 32; ++i) {
    int c = (i * 8 + sub) * 4;
    float4 s = *(float4*)(Srow + c);
    s.x = __expf(s.x - rmax); s.y = __expf(s.y - rmax);
    s.z = __expf(s.z - rmax); s.w = __expf(s.w - rmax);
    *(float4*)(Srow + c) = s;
    lsum += (s.x + s.y) + (s.z + s.w);
  }
  red[row * 8 + sub] = lsum;
  __syncthreads();
  float rsum = 0.0f;
#pragma unroll
  for (int j = 0; j < 8; ++j) rsum += red[row * 8 + j];
  const float inv = 1.0f / rsum;

  bf16* prow = P + ((size_t)b * QLEN + q0 + row) * KVLEN;
  for (int i = 0; i < KVLEN / 32; ++i) {
    int c = (i * 8 + sub) * 4;
    float4 s = *(float4*)(Srow + c);
    v4bf o;
    o[0] = f2bf(s.x * inv); o[1] = f2bf(s.y * inv);
    o[2] = f2bf(s.z * inv); o[3] = f2bf(s.w * inv);
    *(v4bf*)(prow + c) = o;   // packed 8B store
  }
}

extern "C" void kernel_launch(void* const* d_in, const int* in_sizes, int n_in,
                              void* d_out, int out_size, void* d_ws, size_t ws_size,
                              hipStream_t stream) {
  const float* query     = (const float*)d_in[0];
  const float* key_value = (const float*)d_in[1];
  const int*   mask      = (const int*)d_in[2];
  const float* Wq = (const float*)d_in[3];
  const float* bq = (const float*)d_in[4];
  const float* Wk = (const float*)d_in[5];
  const float* bk = (const float*)d_in[6];
  const float* Wv = (const float*)d_in[7];
  const float* bv = (const float*)d_in[8];
  const float* Wo = (const float*)d_in[9];
  const float* bo = (const float*)d_in[10];

  // ---- workspace layout (~200 MB) ----
  // Region 0 (64 MB) is used twice: first as bf16 copies of query/key_value,
  // then (after they are dead) overwritten by P. Deterministic every call.
  char* w = (char*)d_ws;
  bf16* xq_bf  = (bf16*)w;                                  // [B*QL, E]  32MB
  bf16* xkv_bf = (bf16*)(w + (size_t)BSZ * QLEN * EDIM * 2);// [B*KVL,E]  32MB
  bf16* P_bf   = (bf16*)w;                                  // [B,QL,KVL] 64MB (overlay)
  w += (size_t)BSZ * QLEN * KVLEN * 2;
  bf16* Wq_b = (bf16*)w; w += (size_t)EDIM * EDIM * 2;
  bf16* Wk_b = (bf16*)w; w += (size_t)EDIM * EDIM * 2;
  bf16* Wv_b = (bf16*)w; w += (size_t)EDIM * EDIM * 2;
  bf16* Wo_b = (bf16*)w; w += (size_t)EDIM * EDIM * 2;
  bf16* q_bf = (bf16*)w; w += (size_t)BSZ * QLEN * EDIM * 2;   // [B*QL, E]
  bf16* k_bf = (bf16*)w; w += (size_t)BSZ * KVLEN * EDIM * 2;  // [B*KVL, E]
  bf16* vT_b = (bf16*)w; w += (size_t)BSZ * EDIM * KVLEN * 2;  // [B, E, KVL]
  bf16* O_bf = (bf16*)w; w += (size_t)BSZ * QLEN * EDIM * 2;   // [B*QL, E]

  const int nW = EDIM * EDIM;
  const int nX = BSZ * QLEN * EDIM;
  cvt_f32_bf16<<<nW / (256 * 4), 256, 0, stream>>>(Wq, Wq_b, nW);
  cvt_f32_bf16<<<nW / (256 * 4), 256, 0, stream>>>(Wk, Wk_b, nW);
  cvt_f32_bf16<<<nW / (256 * 4), 256, 0, stream>>>(Wv, Wv_b, nW);
  cvt_f32_bf16<<<nW / (256 * 4), 256, 0, stream>>>(Wo, Wo_b, nW);
  cvt_f32_bf16<<<nX / (256 * 4), 256, 0, stream>>>(query, xq_bf, nX);
  cvt_f32_bf16<<<nX / (256 * 4), 256, 0, stream>>>(key_value, xkv_bf, nX);

  dim3 blk(256);
  dim3 gproj(EDIM / 128, (BSZ * QLEN) / 128);  // (8, 128)

  // q = query @ Wq^T + bq
  gemm_nt<<<gproj, blk, 0, stream>>>(xq_bf, EDIM, Wq_b, 0LL, EDIM,
                                     bq, q_bf, 0, EDIM, EDIM, BSZ * QLEN);
  // k = key_value @ Wk^T + bk
  gemm_nt<<<gproj, blk, 0, stream>>>(xkv_bf, EDIM, Wk_b, 0LL, EDIM,
                                     bk, k_bf, 0, EDIM, EDIM, BSZ * KVLEN);
  // v^T[b][e][kv] = (key_value @ Wv^T + bv), transposed per batch
  gemm_nt<<<gproj, blk, 0, stream>>>(xkv_bf, EDIM, Wv_b, 0LL, EDIM,
                                     bv, vT_b, 1, EDIM, EDIM, KVLEN);

  // scores + mask + softmax -> P (bf16); overwrites xq_bf/xkv_bf (dead)
  dim3 gattn(QLEN / 32, BSZ);
  size_t lds = (size_t)32 * KVLEN * 4 + 32 * 8 * 4;  // 257 KB
  attn_scores_softmax<<<gattn, blk, lds, stream>>>(q_bf, k_bf, mask, P_bf,
                                                   1.0f / 32.0f);

  // O = P @ V  ==  P (NT) vT, batched B operand
  gemm_nt<<<gproj, blk, 0, stream>>>(P_bf, KVLEN, vT_b,
                                     (long long)EDIM * KVLEN, KVLEN,
                                     nullptr, O_bf, 0, EDIM, KVLEN, QLEN);

  // out = O @ Wo^T + bo  (f32 store to d_out)
  gemm_nt<<<gproj, blk, 0, stream>>>(O_bf, EDIM, Wo_b, 0LL, EDIM,
                                     bo, d_out, 2, EDIM, EDIM, BSZ * QLEN);
}
